// ViGBlock_7138235646516
// MI455X (gfx1250) — matrix operand order repaired and checked by reading, not compiled, not run on hardware
//
#include <hip/hip_runtime.h>
#include <hip/hip_bf16.h>

// ---------------- problem constants ----------------
#define BB   4
#define CC   192
#define HH   56
#define WWD  56
#define NSP  (HH*WWD)       // 3136
#define TT   (BB*NSP)       // 12544
#define KNN  7
#define C2   (2*CC)         // 384

typedef float    v2f  __attribute__((ext_vector_type(2)));
typedef float    v8f  __attribute__((ext_vector_type(8)));
typedef _Float16 v8h  __attribute__((ext_vector_type(8)));
typedef _Float16 v16h __attribute__((ext_vector_type(16)));

// =====================================================================
// GEMM: A from image layout [b][k][n], W [Cout][K], out seq [T][Cout].
// One wave computes a 16 x (16*CT) output strip: the A fragment is
// loaded once per k-step and reused across CT WMMAs (issue-rate win).
// =====================================================================
template <int CT>
__global__ void gemm_img2seq_kernel(const float* __restrict__ img,
                                    const float* __restrict__ W,
                                    const float* __restrict__ bias,
                                    float* __restrict__ out,
                                    int K, int Cout) {
  int lane = threadIdx.x;
  int m  = lane & 15;
  int hk = lane >> 4;                 // 0/1
  int t0 = blockIdx.x * 16;           // token tile base
  int b  = t0 / NSP;                  // NSP multiple of 16 -> no straddle
  int n0 = t0 - b * NSP;
  int c0 = blockIdx.y * (16 * CT);
  const float* ai = img + (size_t)b * K * NSP + (n0 + m);
  const float* wp[CT];
#pragma unroll
  for (int j = 0; j < CT; ++j) wp[j] = W + (size_t)(c0 + j * 16 + m) * K;
  v8f acc[CT];
#pragma unroll
  for (int j = 0; j < CT; ++j) acc[j] = (v8f){};
  for (int k = 0; k < K; k += 4) {
    int ka = k + hk * 2;
    v2f a;
    a.x = ai[(size_t)ka * NSP];
    a.y = ai[(size_t)(ka + 1) * NSP];
#pragma unroll
    for (int j = 0; j < CT; ++j) {
      v2f bf = *(const v2f*)(wp[j] + ka);
      acc[j] = __builtin_amdgcn_wmma_f32_16x16x4_f32(false, a, false, bf, (short)0, acc[j], false, false);
    }
  }
#pragma unroll
  for (int j = 0; j < CT; ++j) {
    int col = c0 + j * 16 + m;
    float bv = bias[col];
#pragma unroll
    for (int v = 0; v < 8; ++v)
      out[(size_t)(t0 + v + 8 * hk) * Cout + col] = acc[j][v] + bv;
  }
}

// A row-major seq [T][lda], W [Cout][K], out [T][ldo]
template <int CT>
__global__ void gemm_seq_kernel(const float* __restrict__ A, int lda,
                                const float* __restrict__ W,
                                const float* __restrict__ bias,
                                float* __restrict__ out, int ldo, int K) {
  int lane = threadIdx.x;
  int m  = lane & 15;
  int hk = lane >> 4;
  int t0 = blockIdx.x * 16;
  int c0 = blockIdx.y * (16 * CT);
  const float* ap = A + (size_t)(t0 + m) * lda;
  const float* wp[CT];
#pragma unroll
  for (int j = 0; j < CT; ++j) wp[j] = W + (size_t)(c0 + j * 16 + m) * K;
  v8f acc[CT];
#pragma unroll
  for (int j = 0; j < CT; ++j) acc[j] = (v8f){};
  for (int k = 0; k < K; k += 4) {
    int ka = k + hk * 2;
    v2f a = *(const v2f*)(ap + ka);
#pragma unroll
    for (int j = 0; j < CT; ++j) {
      v2f bf = *(const v2f*)(wp[j] + ka);
      acc[j] = __builtin_amdgcn_wmma_f32_16x16x4_f32(false, a, false, bf, (short)0, acc[j], false, false);
    }
  }
#pragma unroll
  for (int j = 0; j < CT; ++j) {
    int col = c0 + j * 16 + m;
    float bv = bias[col];
#pragma unroll
    for (int v = 0; v < 8; ++v)
      out[(size_t)(t0 + v + 8 * hk) * ldo + col] = acc[j][v] + bv;
  }
}

// =====================================================================
// BatchNorm (train) stats -> per-channel scale/shift
// =====================================================================
__global__ void bn_stats_kernel(const float* __restrict__ y,
                                const float* __restrict__ g,
                                const float* __restrict__ bt,
                                float* __restrict__ scale,
                                float* __restrict__ shift, int T, int C) {
  int c = blockIdx.x, tid = threadIdx.x;
  float s = 0.f, s2 = 0.f;
  for (int t = tid; t < T; t += 256) {
    float v = y[(size_t)t * C + c];
    s += v; s2 += v * v;
  }
  __shared__ float rs[256], rq[256];
  rs[tid] = s; rq[tid] = s2; __syncthreads();
  for (int st = 128; st > 0; st >>= 1) {
    if (tid < st) { rs[tid] += rs[tid + st]; rq[tid] += rq[tid + st]; }
    __syncthreads();
  }
  if (tid == 0) {
    float mean = rs[0] / T;
    float var  = rq[0] / T - mean * mean;
    float r    = rsqrtf(var + 1e-5f);
    float sc   = g[c] * r;
    scale[c] = sc;
    shift[c] = bt[c] - mean * sc;
  }
}

__global__ void bn_apply_kernel(const float* __restrict__ y,
                                const float* __restrict__ scale,
                                const float* __restrict__ shift,
                                float* __restrict__ out, int total, int C) {
  int i = blockIdx.x * blockDim.x + threadIdx.x;
  if (i >= total) return;
  int c = i % C;
  out[i] = y[i] * scale[c] + shift[c];
}

// BN apply + residual add + transpose seq->img   out[b][c][n]
__global__ void bn_res_img_kernel(const float* __restrict__ y,
                                  const float* __restrict__ scale,
                                  const float* __restrict__ shift,
                                  const float* __restrict__ res,
                                  float* __restrict__ out) {
  int i = blockIdx.x * blockDim.x + threadIdx.x;
  if (i >= BB * CC * NSP) return;
  int n = i % NSP;
  int c = (i / NSP) % CC;
  int b = i / (NSP * CC);
  float v = y[(size_t)(b * NSP + n) * CC + c] * scale[c] + shift[c];
  out[i] = v + res[i];
}

// =====================================================================
// Row L2-normalize -> f16 for WMMA + squared norm of normalized row
// =====================================================================
__global__ void rownorm_kernel(const float* __restrict__ h,
                               _Float16* __restrict__ xnh,
                               float* __restrict__ sq) {
  int t = blockIdx.x, c = threadIdx.x;     // 192 threads
  float v = h[(size_t)t * CC + c];
  __shared__ float red[256];
  __shared__ float invs;
  red[c] = v * v;
  if (c < 64) red[192 + c] = 0.f;
  __syncthreads();
  for (int st = 128; st > 0; st >>= 1) {
    if (c < st) red[c] += red[c + st];
    __syncthreads();
  }
  if (c == 0) {
    float nrm = sqrtf(red[0]);
    invs = 1.f / fmaxf(nrm, 1e-12f);
    sq[t] = red[0] * invs * invs;
  }
  __syncthreads();
  xnh[(size_t)t * CC + c] = (_Float16)(v * invs);
}

// =====================================================================
// Fused KNN: f16 WMMA similarity + rel_pos + streaming per-row top-7
// one wave per 16-row tile; never materializes the 3136x3136 matrix
// =====================================================================
__global__ void knn_kernel(const _Float16* __restrict__ xnh,
                           const float* __restrict__ sq,
                           const float* __restrict__ relpos,
                           int* __restrict__ idx) {
  const int JT = NSP / 16;                // 196
  int lane = threadIdx.x;
  int m  = lane & 15;
  int hk = lane >> 4;
  int b  = blockIdx.x / JT;
  int i0 = (blockIdx.x % JT) * 16;

  // preload A fragments for the 16 rows (K = 192 -> 6 wmma steps)
  v16h afr[6];
  const _Float16* arow = xnh + (size_t)(b * NSP + i0 + m) * CC;
#pragma unroll
  for (int s = 0; s < 6; ++s) {
    v8h lo = *(const v8h*)(arow + s * 32 + hk * 8);
    v8h hi = *(const v8h*)(arow + s * 32 + hk * 8 + 16);
    v16h a;
#pragma unroll
    for (int j = 0; j < 8; ++j) { a[j] = lo[j]; a[8 + j] = hi[j]; }
    afr[s] = a;
  }

  float tv[KNN]; int ti[KNN];             // sorted ascending, tv[0] = min
#pragma unroll
  for (int q = 0; q < KNN; ++q) { tv[q] = -3.0e38f; ti[q] = 0; }

  __shared__ float smem[16][17];

  for (int jt = 0; jt < JT; ++jt) {
    int j0 = jt * 16;
    const _Float16* brow = xnh + (size_t)(b * NSP + j0 + m) * CC;
    // prefetch next tile's B rows and rel_pos strip (global_prefetch_b8)
    if (jt + 1 < JT) {
      __builtin_prefetch(xnh + (size_t)(b * NSP + j0 + 16 + m) * CC, 0, 1);
      __builtin_prefetch(relpos + (size_t)(i0 + m) * NSP + (j0 + 16), 0, 1);
    }
    v8f acc = {};
#pragma unroll
    for (int s = 0; s < 6; ++s) {
      v8h blo = *(const v8h*)(brow + s * 32 + hk * 8);
      v8h bhi = *(const v8h*)(brow + s * 32 + hk * 8 + 16);
      v16h bf;
#pragma unroll
      for (int j = 0; j < 8; ++j) { bf[j] = blo[j]; bf[8 + j] = bhi[j]; }
      acc = __builtin_amdgcn_wmma_f32_16x16x32_f16(false, afr[s], false, bf, (short)0, acc, false, false);
    }
    float sqn = sq[b * NSP + j0 + m];     // column n = m
#pragma unroll
    for (int v = 0; v < 8; ++v) {
      int row = v + 8 * hk;
      float sc = 2.f * acc[v] - sqn - relpos[(size_t)(i0 + row) * NSP + (j0 + m)];
      smem[row][m] = sc;
    }
    __syncthreads();
    // lane pair (m, m+16) own row m; each merges 8 candidates
#pragma unroll
    for (int i = 0; i < 8; ++i) {
      int n = hk * 8 + i;
      float c = smem[m][n];
      if (c > tv[0]) {
        int p = 0, jj = j0 + n;
#pragma unroll
        for (int q = 1; q < KNN; ++q) {
          if (c > tv[q]) { tv[q - 1] = tv[q]; ti[q - 1] = ti[q]; p = q; }
        }
        tv[p] = c; ti[p] = jj;
      }
    }
    __syncthreads();
  }

  __shared__ float sv[32][KNN];
  __shared__ int   si[32][KNN];
#pragma unroll
  for (int q = 0; q < KNN; ++q) { sv[lane][q] = tv[q]; si[lane][q] = ti[q]; }
  __syncthreads();
  if (lane < 16) {
    int ia = KNN - 1, ib = KNN - 1;
    int* op = idx + (size_t)(b * NSP + i0 + lane) * KNN;
#pragma unroll
    for (int k = 0; k < KNN; ++k) {
      float va = sv[lane][ia], vb = sv[lane + 16][ib];
      int   ja = si[lane][ia], jb = si[lane + 16][ib];
      bool ta = (va > vb) || (va == vb && ja <= jb);
      if (ta) { op[k] = ja; --ia; } else { op[k] = jb; --ib; }
    }
  }
}

// =====================================================================
// gather neighbors + max-relative + interleave -> z[t][2c]=h, [2c+1]=mr
// =====================================================================
__global__ void gather_z_kernel(const float* __restrict__ h,
                                const int* __restrict__ idx,
                                float* __restrict__ z) {
  int t = blockIdx.x, c = threadIdx.x;  // 192 threads
  __shared__ int nb[KNN];
  if (c < KNN) nb[c] = idx[(size_t)t * KNN + c];
  __syncthreads();
  int b = t / NSP;
  float hv = h[(size_t)t * CC + c];
  float mx = -3.0e38f;
#pragma unroll
  for (int k = 0; k < KNN; ++k)
    mx = fmaxf(mx, h[((size_t)b * NSP + nb[k]) * CC + c]);
  z[(size_t)t * C2 + 2 * c]     = hv;
  z[(size_t)t * C2 + 2 * c + 1] = mx - hv;
}

// =====================================================================
// instance norm over sequence (per b,channel) + exact GELU, in place
// =====================================================================
__global__ void inorm_gelu_kernel(float* __restrict__ z2) {
  int blk = blockIdx.x;              // b*384 + ch
  int b = blk / C2, ch = blk % C2;
  int tid = threadIdx.x;
  const size_t base = (size_t)b * NSP * C2 + ch;
  float s = 0.f, s2 = 0.f;
  for (int n = tid; n < NSP; n += 256) {
    float v = z2[base + (size_t)n * C2];
    s += v; s2 += v * v;
  }
  __shared__ float rs[256], rq[256];
  rs[tid] = s; rq[tid] = s2; __syncthreads();
  for (int st = 128; st > 0; st >>= 1) {
    if (tid < st) { rs[tid] += rs[tid + st]; rq[tid] += rq[tid + st]; }
    __syncthreads();
  }
  float mean = rs[0] / NSP;
  float var  = rq[0] / NSP - mean * mean;
  float r    = rsqrtf(var + 1e-5f);
  for (int n = tid; n < NSP; n += 256) {
    size_t ix = base + (size_t)n * C2;
    float v = (z2[ix] - mean) * r;
    z2[ix] = 0.5f * v * (1.0f + erff(v * 0.70710678118654752f));
  }
}

// =====================================================================
// SRU
// =====================================================================
__global__ void sum_gamma_kernel(const float* __restrict__ g, float* __restrict__ out) {
  int tid = threadIdx.x;               // 192
  __shared__ float red[256];
  red[tid] = g[tid];
  if (tid < 64) red[192 + tid] = 0.f;
  __syncthreads();
  for (int st = 128; st > 0; st >>= 1) {
    if (tid < st) red[tid] += red[tid + st];
    __syncthreads();
  }
  if (tid == 0) out[0] = red[0];
}

__global__ void sru_stats_kernel(const float* __restrict__ x,
                                 float* __restrict__ smean,
                                 float* __restrict__ sinv) {
  int blk = blockIdx.x;                // b*4+g
  int b = blk >> 2, g = blk & 3;
  const int cnt = (CC / 4) * NSP;      // 150528
  const float* p = x + (size_t)b * CC * NSP + (size_t)g * (CC / 4) * NSP;
  int tid = threadIdx.x;
  float s = 0.f, s2 = 0.f;
  for (int i = tid; i < cnt; i += 256) {
    float v = p[i]; s += v; s2 += v * v;
  }
  __shared__ float rs[256], rq[256];
  rs[tid] = s; rq[tid] = s2; __syncthreads();
  for (int st = 128; st > 0; st >>= 1) {
    if (tid < st) { rs[tid] += rs[tid + st]; rq[tid] += rq[tid + st]; }
    __syncthreads();
  }
  if (tid == 0) {
    float mean = rs[0] / cnt;
    float var1 = (rq[0] - cnt * mean * mean) / (cnt - 1);   // ddof=1
    float sd   = sqrtf(fmaxf(var1, 0.f));
    smean[blk] = mean;
    sinv[blk]  = 1.f / (sd + 1e-10f);
  }
}

__global__ void sru_apply_kernel(const float* __restrict__ x,
                                 const float* __restrict__ gamma,
                                 const float* __restrict__ beta,
                                 const float* __restrict__ smean,
                                 const float* __restrict__ sinv,
                                 const float* __restrict__ gsum,
                                 float* __restrict__ out) {
  int i = blockIdx.x * blockDim.x + threadIdx.x;
  const int total = BB * (CC / 2) * NSP;
  if (i >= total) return;
  int n = i % NSP;
  int c = (i / NSP) % (CC / 2);
  int b = i / (NSP * (CC / 2));
  int c2 = c + CC / 2;
  float gs = gsum[0];
  size_t i1 = (size_t)b * CC * NSP + (size_t)c  * NSP + n;
  size_t i2 = (size_t)b * CC * NSP + (size_t)c2 * NSP + n;
  float x1 = x[i1], x2 = x[i2];
  int g1 = c / 48, g2 = c2 / 48;
  float gn1 = (x1 - smean[b * 4 + g1]) * sinv[b * 4 + g1] * gamma[c]  + beta[c];
  float gn2 = (x2 - smean[b * 4 + g2]) * sinv[b * 4 + g2] * gamma[c2] + beta[c2];
  bool m1 = (gn1 * (gamma[c]  / gs)) >= 0.f;   // sigmoid(t)>=0.5 <=> t>=0
  bool m2 = (gn2 * (gamma[c2] / gs)) >= 0.f;
  out[i1] = (m1 ? x1 : 0.f) + (m2 ? 0.f : x2);
  out[i2] = (m2 ? x2 : 0.f) + (m1 ? 0.f : x1);
}

// =====================================================================
// CRU
// =====================================================================
__global__ void cru_uplow_kernel(const float* __restrict__ xs,
                                 const float* __restrict__ w1,
                                 const float* __restrict__ w2,
                                 float* __restrict__ up,
                                 float* __restrict__ low) {
  int i = blockIdx.x * blockDim.x + threadIdx.x;
  const int half_total = BB * 48 * NSP;
  if (i >= 2 * half_total) return;
  int hsel = i / half_total;
  int r = i - hsel * half_total;
  int n = r % NSP;
  int u = (r / NSP) % 48;
  int b = r / (48 * NSP);
  const float* w = hsel ? w2 : w1;
  const float* xin = xs + (size_t)b * CC * NSP + (size_t)(hsel ? 96 : 0) * NSP;
  float acc = 0.f;
  for (int k = 0; k < 96; ++k) acc += xin[(size_t)k * NSP + n] * w[u * 96 + k];
  (hsel ? low : up)[r] = acc;
}

__global__ void cru_y1_kernel(const float* __restrict__ up,
                              const float* __restrict__ gwc_w,
                              const float* __restrict__ gwc_b,
                              const float* __restrict__ pwc1_w,
                              float* __restrict__ y1) {
  int i = blockIdx.x * blockDim.x + threadIdx.x;
  if (i >= BB * CC * NSP) return;
  int n = i % NSP;
  int o = (i / NSP) % CC;
  int b = i / (CC * NSP);
  int h = n / WWD, w = n % WWD;
  int gsel = o / 96;                    // feature_group_count=2
  const float* ub = up + (size_t)b * 48 * NSP;
  float acc = gwc_b[o];
  for (int kc = 0; kc < 24; ++kc) {
    int ic = gsel * 24 + kc;
    const float* uc = ub + (size_t)ic * NSP;
    const float* wp = gwc_w + ((size_t)o * 24 + kc) * 9;
#pragma unroll
    for (int dy = -1; dy <= 1; ++dy) {
      int hh = h + dy;
      if (hh < 0 || hh >= HH) continue;
#pragma unroll
      for (int dx = -1; dx <= 1; ++dx) {
        int ww = w + dx;
        if (ww < 0 || ww >= WWD) continue;
        acc += uc[hh * WWD + ww] * wp[(dy + 1) * 3 + (dx + 1)];
      }
    }
  }
  for (int kc = 0; kc < 48; ++kc)
    acc += ub[(size_t)kc * NSP + n] * pwc1_w[o * 48 + kc];
  y1[i] = acc;
}

__global__ void cru_y2_kernel(const float* __restrict__ low,
                              const float* __restrict__ pwc2_w,
                              float* __restrict__ y2) {
  int i = blockIdx.x * blockDim.x + threadIdx.x;
  if (i >= BB * CC * NSP) return;
  int n = i % NSP;
  int o = (i / NSP) % CC;
  int b = i / (CC * NSP);
  const float* lb = low + (size_t)b * 48 * NSP;
  float acc;
  if (o < 144) {
    acc = 0.f;
    for (int k = 0; k < 48; ++k) acc += lb[(size_t)k * NSP + n] * pwc2_w[o * 48 + k];
  } else {
    acc = lb[(size_t)(o - 144) * NSP + n];
  }
  y2[i] = acc;
}

__global__ void chmean_kernel(const float* __restrict__ y1,
                              const float* __restrict__ y2,
                              float* __restrict__ cm) {
  int blk = blockIdx.x;                 // b*384 + ch
  int b = blk / C2, ch = blk % C2;
  const float* src = (ch < CC) ? (y1 + (size_t)b * CC * NSP + (size_t)ch * NSP)
                               : (y2 + (size_t)b * CC * NSP + (size_t)(ch - CC) * NSP);
  int tid = threadIdx.x;
  float s = 0.f;
  for (int n = tid; n < NSP; n += 256) s += src[n];
  __shared__ float red[256];
  red[tid] = s; __syncthreads();
  for (int st = 128; st > 0; st >>= 1) {
    if (tid < st) red[tid] += red[tid + st];
    __syncthreads();
  }
  if (tid == 0) cm[blk] = red[0] / NSP;
}

__global__ void softmax384_kernel(const float* __restrict__ cm, float* __restrict__ sft) {
  int b = blockIdx.x, tid = threadIdx.x;   // 384 threads
  float v = cm[b * C2 + tid];
  __shared__ float red[512];
  red[tid] = v;
  if (tid < 128) red[C2 + tid] = -3.0e38f;
  __syncthreads();
  for (int st = 256; st > 0; st >>= 1) {
    if (tid < st) red[tid] = fmaxf(red[tid], red[tid + st]);
    __syncthreads();
  }
  float mx = red[0];
  __syncthreads();
  float e = expf(v - mx);
  red[tid] = e;
  if (tid < 128) red[C2 + tid] = 0.f;
  __syncthreads();
  for (int st = 256; st > 0; st >>= 1) {
    if (tid < st) red[tid] += red[tid + st];
    __syncthreads();
  }
  sft[b * C2 + tid] = e / red[0];
}

__global__ void cru_combine_kernel(const float* __restrict__ y1,
                                   const float* __restrict__ y2,
                                   const float* __restrict__ sft,
                                   float* __restrict__ out) {
  int i = blockIdx.x * blockDim.x + threadIdx.x;
  if (i >= BB * CC * NSP) return;
  int o = (i / NSP) % CC;
  int b = i / (CC * NSP);
  out[i] = y1[i] * sft[b * C2 + o] + y2[i] * sft[b * C2 + CC + o];
}

// =====================================================================
// instance norm NCHW
// =====================================================================
__global__ void inorm_img_stats_kernel(const float* __restrict__ x,
                                       float* __restrict__ mean,
                                       float* __restrict__ rstd) {
  int blk = blockIdx.x;                  // b*CC + c
  const float* p = x + (size_t)blk * NSP;
  int tid = threadIdx.x;
  float s = 0.f, s2 = 0.f;
  for (int n = tid; n < NSP; n += 256) {
    float v = p[n]; s += v; s2 += v * v;
  }
  __shared__ float rs[256], rq[256];
  rs[tid] = s; rq[tid] = s2; __syncthreads();
  for (int st = 128; st > 0; st >>= 1) {
    if (tid < st) { rs[tid] += rs[tid + st]; rq[tid] += rq[tid + st]; }
    __syncthreads();
  }
  if (tid == 0) {
    float mn = rs[0] / NSP;
    float var = rq[0] / NSP - mn * mn;
    mean[blk] = mn;
    rstd[blk] = rsqrtf(var + 1e-5f);
  }
}

__global__ void inorm_img_apply_kernel(const float* __restrict__ x,
                                       const float* __restrict__ mean,
                                       const float* __restrict__ rstd,
                                       const float* __restrict__ res,
                                       float* __restrict__ out, int do_relu) {
  int i = blockIdx.x * blockDim.x + threadIdx.x;
  if (i >= BB * CC * NSP) return;
  int bc = i / NSP;
  float v = (x[i] - mean[bc]) * rstd[bc];
  if (do_relu) v = fmaxf(v, 0.f);
  if (res) v += res[i];
  out[i] = v;
}

// =====================================================================
// host orchestration
// =====================================================================
struct GP { const float *fc1_w,*fc1_b,*fc1_g,*fc1_bt,*gc_w,*gc_b,*fc2_w,*fc2_b,*fc2_g,*fc2_bt; };
struct SP { const float *sru_g,*sru_b,*sq1_w,*sq2_w,*gwc_w,*gwc_b,*pwc1_w,*pwc2_w; };

static void run_grapher(const float* in_img, const GP& P, float* out_img, const float* relpos,
                        float* ybuf, float* hbuf, _Float16* xnh, float* sqb, int* idxb,
                        float* scaleb, float* shiftb, float* z, float* z2, hipStream_t st) {
  // fc1: 16x64 strip per wave (Cout=192 -> 3 col blocks)
  gemm_img2seq_kernel<4><<<dim3(TT/16, 3), 32, 0, st>>>(in_img, P.fc1_w, P.fc1_b, ybuf, CC, CC);
  bn_stats_kernel<<<CC, 256, 0, st>>>(ybuf, P.fc1_g, P.fc1_bt, scaleb, shiftb, TT, CC);
  bn_apply_kernel<<<(TT*CC+255)/256, 256, 0, st>>>(ybuf, scaleb, shiftb, hbuf, TT*CC, CC);
  rownorm_kernel<<<TT, 192, 0, st>>>(hbuf, xnh, sqb);
  knn_kernel<<<BB*(NSP/16), 32, 0, st>>>(xnh, sqb, relpos, idxb);
  gather_z_kernel<<<TT, 192, 0, st>>>(hbuf, idxb, z);
  // gconv: per group Cout=96 -> 16x48 strip per wave, 2 col blocks
  for (int g = 0; g < 4; ++g)
    gemm_seq_kernel<3><<<dim3(TT/16, 2), 32, 0, st>>>(z + g*96, C2, P.gc_w + (size_t)g*96*96,
                                                      P.gc_b + g*96, z2 + g*96, C2, 96);
  inorm_gelu_kernel<<<BB*C2, 256, 0, st>>>(z2);
  // fc2: 16x64 strip per wave
  gemm_seq_kernel<4><<<dim3(TT/16, 3), 32, 0, st>>>(z2, C2, P.fc2_w, P.fc2_b, ybuf, CC, C2);
  bn_stats_kernel<<<CC, 256, 0, st>>>(ybuf, P.fc2_g, P.fc2_bt, scaleb, shiftb, TT, CC);
  bn_res_img_kernel<<<(BB*CC*NSP+255)/256, 256, 0, st>>>(ybuf, scaleb, shiftb, in_img, out_img);
}

static void run_scconv(const float* in_img, const SP& S, float* out_img,
                       float* sruout, float* up, float* low, float* y1, float* y2,
                       float* smean, float* sinv, float* gsum, float* cm, float* sft,
                       hipStream_t st) {
  sum_gamma_kernel<<<1, 192, 0, st>>>(S.sru_g, gsum);
  sru_stats_kernel<<<BB*4, 256, 0, st>>>(in_img, smean, sinv);
  sru_apply_kernel<<<(BB*96*NSP+255)/256, 256, 0, st>>>(in_img, S.sru_g, S.sru_b, smean, sinv, gsum, sruout);
  cru_uplow_kernel<<<(2*BB*48*NSP+255)/256, 256, 0, st>>>(sruout, S.sq1_w, S.sq2_w, up, low);
  cru_y1_kernel<<<(BB*CC*NSP+255)/256, 256, 0, st>>>(up, S.gwc_w, S.gwc_b, S.pwc1_w, y1);
  cru_y2_kernel<<<(BB*CC*NSP+255)/256, 256, 0, st>>>(low, S.pwc2_w, y2);
  chmean_kernel<<<BB*C2, 256, 0, st>>>(y1, y2, cm);
  softmax384_kernel<<<BB, C2, 0, st>>>(cm, sft);
  cru_combine_kernel<<<(BB*CC*NSP+255)/256, 256, 0, st>>>(y1, y2, sft, out_img);
}

extern "C" void kernel_launch(void* const* d_in, const int* in_sizes, int n_in,
                              void* d_out, int out_size, void* d_ws, size_t ws_size,
                              hipStream_t stream) {
  (void)n_in; (void)out_size; (void)ws_size;
  const float* x = (const float*)d_in[0];
  const float* relpos;
  GP g1, g2; SP s1, s2;
  auto F = [&](int i) { return (const float*)d_in[i]; };

  if (in_sizes[1] == CC * CC) {
    // pytree preserved insertion order
    g1 = { F(1),F(2),F(3),F(4),F(5),F(6),F(7),F(8),F(9),F(10) };
    s1 = { F(11),F(12),F(13),F(14),F(15),F(16),F(17),F(18) };
    g2 = { F(19),F(20),F(21),F(22),F(23),F(24),F(25),F(26),F(27),F(28) };
    s2 = { F(29),F(30),F(31),F(32),F(33),F(34),F(35),F(36) };
  } else {
    // JAX sorted-key flattening: g1,g2,s1,s2; within grapher: fc1_b,fc1_bt,fc1_g,fc1_w,
    // fc2_b,fc2_bt,fc2_g,fc2_w,gc_b,gc_w; scconv: gwc_b,gwc_w,pwc1_w,pwc2_w,sq1_w,sq2_w,sru_b,sru_g
    g1 = { F(4),F(1),F(3),F(2),F(10),F(9),F(8),F(5),F(7),F(6) };
    g2 = { F(14),F(11),F(13),F(12),F(20),F(19),F(18),F(15),F(17),F(16) };
    s1 = { F(28),F(27),F(25),F(26),F(22),F(21),F(23),F(24) };
    s2 = { F(36),F(35),F(33),F(34),F(30),F(29),F(31),F(32) };
  }
  relpos = F(37);

  // ---- workspace carve-up (all 256B aligned) ----
  char* wsp = (char*)d_ws;
  size_t off = 0;
  auto take = [&](size_t bytes) -> void* {
    void* p = wsp + off;
    off = (off + bytes + 255) & ~(size_t)255;
    return p;
  };
  const size_t SZ_SEQ = (size_t)TT * CC * sizeof(float);        // 9.63 MB
  const size_t SZ_Z   = (size_t)TT * C2 * sizeof(float);        // 19.27 MB

  float* bufA   = (float*)take(SZ_SEQ);   // fc out / cru out
  float* bufB   = (float*)take(SZ_SEQ);   // h (bn-applied)
  float* bufC   = (float*)take(SZ_SEQ);   // grapher out img / relu-inorm img
  _Float16* xnh = (_Float16*)take((size_t)TT * CC * sizeof(_Float16));
  float* sqb    = (float*)take((size_t)TT * sizeof(float));
  int*   idxb   = (int*)  take((size_t)TT * KNN * sizeof(int));
  float* scaleb = (float*)take(CC * sizeof(float));
  float* shiftb = (float*)take(CC * sizeof(float));
  float* smean  = (float*)take(BB * 4 * sizeof(float));
  float* sinv   = (float*)take(BB * 4 * sizeof(float));
  float* gsum   = (float*)take(256);
  float* cm     = (float*)take(BB * C2 * sizeof(float));
  float* sft    = (float*)take(BB * C2 * sizeof(float));
  float* imean  = (float*)take(BB * CC * sizeof(float));
  float* irstd  = (float*)take(BB * CC * sizeof(float));
  float* z      = (float*)take(SZ_Z);
  float* z2     = (float*)take(SZ_Z);     // contiguous after z

  // scconv temporaries alias the (then unused) z/z2 region
  float* sruout = z;
  float* up     = z + (size_t)BB * CC * NSP;                         // after sruout
  float* low    = up + (size_t)BB * 48 * NSP;
  float* y1     = low + (size_t)BB * 48 * NSP;
  float* y2     = y1 + (size_t)BB * CC * NSP;                        // spills into z2 region (free)

  const int NE = BB * CC * NSP;

  // ----- block 1: grapher(g1) -> scconv(s1) -> relu(inorm) -----
  run_grapher(x, g1, bufC, relpos, bufA, bufB, xnh, sqb, idxb, scaleb, shiftb, z, z2, stream);
  run_scconv(bufC, s1, bufA, sruout, up, low, y1, y2, smean, sinv, gsum, cm, sft, stream);
  inorm_img_stats_kernel<<<BB*CC, 256, 0, stream>>>(bufA, imean, irstd);
  inorm_img_apply_kernel<<<(NE+255)/256, 256, 0, stream>>>(bufA, imean, irstd, nullptr, bufC, 1);

  // ----- block 2: grapher(g2) -> scconv(s2) -> inorm, residual with x -----
  run_grapher(bufC, g2, bufC, relpos, bufA, bufB, xnh, sqb, idxb, scaleb, shiftb, z, z2, stream);
  run_scconv(bufC, s2, bufA, sruout, up, low, y1, y2, smean, sinv, gsum, cm, sft, stream);
  inorm_img_stats_kernel<<<BB*CC, 256, 0, stream>>>(bufA, imean, irstd);
  inorm_img_apply_kernel<<<(NE+255)/256, 256, 0, stream>>>(bufA, imean, irstd, x, (float*)d_out, 0);
}
